// Style2ResidualBlock1DSrc_54966991454473
// MI455X (gfx1250) — matrix operand order repaired and checked by reading, not compiled, not run on hardware
//
#include <hip/hip_runtime.h>
#include <math.h>

#define DIN   512
#define DOUT  512
#define TLEN  2048
#define NB    16
#define KW    3

typedef __bf16 bf16;
typedef __attribute__((ext_vector_type(4)))  __bf16 bf16x4;
typedef __attribute__((ext_vector_type(8)))  __bf16 bf16x8;
typedef __attribute__((ext_vector_type(16))) __bf16 v16bf;
typedef __attribute__((ext_vector_type(8)))  float  v8f;
typedef __attribute__((ext_vector_type(4)))  float  f32x4;

// LDS row padding: 40 bf16 = 80 bytes (multiple of 16B, spreads banks)
#define WROW 40
#define XROW 40

// ---------------------------------------------------------------------------
// Phase 0: x fp32 [b][i][t] -> bf16 transposed [b][t][i]
// One block per 32x32 tile; LDS-tiled transpose, coalesced both sides.
// ---------------------------------------------------------------------------
__global__ __launch_bounds__(256) void xpose_bf16(const float* __restrict__ x,
                                                  bf16* __restrict__ xT) {
    const int b  = blockIdx.z;
    const int i0 = blockIdx.y * 32;
    const int t0 = blockIdx.x * 32;
    const int tid = threadIdx.x;

    __shared__ float tile[32][33];

    // read 32(i) x 32(t), coalesced along t: thread -> row r, 4 cols
    {
        const int r  = tid >> 3;          // 0..31 (i)
        const int c4 = (tid & 7) * 4;     // 0,4,..,28 (t)
        f32x4 v = *(const f32x4*)&x[((size_t)b * DIN + i0 + r) * TLEN + t0 + c4];
        tile[r][c4 + 0] = v[0];
        tile[r][c4 + 1] = v[1];
        tile[r][c4 + 2] = v[2];
        tile[r][c4 + 3] = v[3];
    }
    __syncthreads();
    // write 32(t) x 32(i), coalesced along i
    {
        const int r  = tid >> 3;          // 0..31 (t)
        const int c4 = (tid & 7) * 4;     // 0,4,..,28 (i)
        bf16x4 o;
        o[0] = (bf16)tile[c4 + 0][r];
        o[1] = (bf16)tile[c4 + 1][r];
        o[2] = (bf16)tile[c4 + 2][r];
        o[3] = (bf16)tile[c4 + 3][r];
        *(bf16x4*)&xT[((size_t)b * TLEN + t0 + r) * DIN + i0 + c4] = o;
    }
}

// ---------------------------------------------------------------------------
// Phase 1: style code  s[b,i] = (1/16) * sum_j c[b,j]*style_w[i,j] + style_b[i]
// ---------------------------------------------------------------------------
__global__ __launch_bounds__(256) void style_kernel(const float* __restrict__ c_src,
                                                    const float* __restrict__ c_trg,
                                                    const float* __restrict__ sw,
                                                    const float* __restrict__ sb,
                                                    float* __restrict__ s) {
    const int b = blockIdx.y;
    const int i = blockIdx.x * 256 + threadIdx.x;      // 0..511
    const float* row = sw + (size_t)i * 256;
    float acc = 0.f;
    #pragma unroll 4
    for (int j = 0; j < 128; ++j) acc += c_src[b * 128 + j] * row[j];
    #pragma unroll 4
    for (int j = 0; j < 128; ++j) acc += c_trg[b * 128 + j] * row[128 + j];
    s[(size_t)b * DIN + i] = acc * 0.0625f + sb[i];    // LIN_SCALE = 1/sqrt(256)
}

// ---------------------------------------------------------------------------
// Phase 2: modulate + demodulate weights, write bf16 to ws as [b][k][O][I]
// one block of 128 threads per (b, o)
// ---------------------------------------------------------------------------
__global__ __launch_bounds__(128) void modw_kernel(const float* __restrict__ weight,
                                                   const float* __restrict__ s,
                                                   bf16* __restrict__ wb) {
    const int o = blockIdx.x;
    const int b = blockIdx.y;
    const int tid = threadIdx.x;
    const float CONV_SCALE = 0.014731391274719742f;    // 1/sqrt(512*9)

    float wv[4][KW];
    float loc = 0.f;
    #pragma unroll
    for (int u = 0; u < 4; ++u) {
        const int i = tid + u * 128;
        const float si = s[(size_t)b * DIN + i] * CONV_SCALE;
        #pragma unroll
        for (int k = 0; k < KW; ++k) {
            float v = weight[((size_t)o * DIN + i) * KW + k] * si;
            wv[u][k] = v;
            loc += v * v;
        }
    }
    __shared__ float red[128];
    red[tid] = loc;
    __syncthreads();
    for (int st = 64; st > 0; st >>= 1) {
        if (tid < st) red[tid] += red[tid + st];
        __syncthreads();
    }
    const float demod = rsqrtf(red[0] + 1e-8f);
    #pragma unroll
    for (int u = 0; u < 4; ++u) {
        const int i = tid + u * 128;
        #pragma unroll
        for (int k = 0; k < KW; ++k) {
            wb[(((size_t)b * KW + k) * DOUT + o) * DIN + i] = (bf16)(wv[u][k] * demod);
        }
    }
}

// ---------------------------------------------------------------------------
// Phase 3: conv1d as 3 shifted bf16 WMMA GEMMs with 2x4 register blocking.
// Block = 256 threads (8 waves = 4 O-waves x 2 T-waves).
// Each wave: 2 O-subtiles x 4 T-subtiles (32x64 out) -> block tile 128(O)x128(T).
// ---------------------------------------------------------------------------
__global__ __launch_bounds__(256) void conv_wmma(const bf16* __restrict__ xT,
                                                 const bf16* __restrict__ wb,
                                                 float* __restrict__ out) {
    const int b    = blockIdx.z;
    const int o0   = blockIdx.y * 128;
    const int t0   = blockIdx.x * 128;
    const int tid  = threadIdx.x;
    const int lane = tid & 31;
    const int wave = tid >> 5;
    const int ow   = wave >> 1;         // 0..3
    const int tw   = wave & 1;          // 0..1
    const int half = lane >> 4;         // 0/1
    const int l15  = lane & 15;

    __shared__ bf16 sW[KW * 128 * WROW];   // [k][o(128)][i(32)] padded rows
    __shared__ bf16 sXT[130 * XROW];       // [t_local(130)][i(32)] padded rows

    v8f acc[2][4];
    #pragma unroll
    for (int so = 0; so < 2; ++so)
        #pragma unroll
        for (int st = 0; st < 4; ++st)
            acc[so][st] = (v8f){0.f, 0.f, 0.f, 0.f, 0.f, 0.f, 0.f, 0.f};

    const size_t xbase = (size_t)b * TLEN * DIN;     // xT: [t][i]
    const size_t wbase = (size_t)b * KW * DOUT * DIN;

    for (int i0 = 0; i0 < DIN; i0 += 32) {
        __syncthreads();
        // ---- stage weights: 3*128*32 = 12288 bf16, 6 x vec8 per thread ----
        #pragma unroll
        for (int r = 0; r < 6; ++r) {
            const int e   = (tid + r * 256) * 8;   // logical element in [0,12288)
            const int kk  = e >> 12;               // /4096
            const int rem = e & 4095;
            const int oo  = rem >> 5;
            const int ii  = rem & 31;
            *(bf16x8*)&sW[(kk * 128 + oo) * WROW + ii] =
                *(const bf16x8*)&wb[wbase + ((size_t)kk * DOUT + o0 + oo) * DIN + i0 + ii];
        }
        // ---- stage xT rows t0-1 .. t0+128 (130 rows x 32 i), vec8 loads ----
        for (int idx = tid; idx < 130 * 4; idx += 256) {
            const int tl = idx >> 2;
            const int q  = (idx & 3) * 8;
            const int gt = t0 - 1 + tl;
            bf16x8 v;
            if (gt >= 0 && gt < TLEN) {
                v = *(const bf16x8*)&xT[xbase + (size_t)gt * DIN + i0 + q];
            } else {
                #pragma unroll
                for (int j = 0; j < 8; ++j) v[j] = (bf16)0.f;
            }
            *(bf16x8*)&sXT[tl * XROW + q] = v;
        }
        // hint next weight chunk toward the caches (global_prefetch_b8)
        if (i0 + 32 < DIN)
            __builtin_prefetch(&wb[wbase + (size_t)(o0 + (tid >> 1)) * DIN + i0 + 32], 0, 0);
        __syncthreads();

        // ---- 3 taps x (2 O-subtiles x 4 T-subtiles) WMMAs ----
        #pragma unroll
        for (int k = 0; k < KW; ++k) {
            // A fragments (weights): lane row = l15, K runs [8h..+7],[16+8h..+7]
            v16bf A[2];
            #pragma unroll
            for (int so = 0; so < 2; ++so) {
                const bf16* wrow = &sW[(k * 128 + ow * 32 + so * 16 + l15) * WROW];
                bf16x8 alo = *(const bf16x8*)&wrow[8 * half];
                bf16x8 ahi = *(const bf16x8*)&wrow[16 + 8 * half];
                A[so] = __builtin_shufflevector(alo, ahi,
                            0, 1, 2, 3, 4, 5, 6, 7, 8, 9, 10, 11, 12, 13, 14, 15);
            }
            // B fragments (x^T): lane col = l15 (t), K runs [8h..+7],[16+8h..+7]
            v16bf Bf[4];
            #pragma unroll
            for (int st = 0; st < 4; ++st) {
                const int tl = tw * 64 + st * 16 + l15 + k;   // maps to t0-1+tl
                const bf16* xcol = &sXT[tl * XROW];
                bf16x8 blo = *(const bf16x8*)&xcol[8 * half];
                bf16x8 bhi = *(const bf16x8*)&xcol[16 + 8 * half];
                Bf[st] = __builtin_shufflevector(blo, bhi,
                            0, 1, 2, 3, 4, 5, 6, 7, 8, 9, 10, 11, 12, 13, 14, 15);
            }
            #pragma unroll
            for (int so = 0; so < 2; ++so)
                #pragma unroll
                for (int st = 0; st < 4; ++st)
                    acc[so][st] = __builtin_amdgcn_wmma_f32_16x16x32_bf16(
                        false, A[so], false, Bf[st], (short)0, acc[so][st], false, false);
        }
    }

    // ---- store D: VGPR r -> M = 8*half + r, N = l15 ----
    #pragma unroll
    for (int so = 0; so < 2; ++so) {
        const int orow = o0 + ow * 32 + so * 16 + 8 * half;
        #pragma unroll
        for (int st = 0; st < 4; ++st) {
            const int tcol = t0 + tw * 64 + st * 16 + l15;
            float* obase = out + ((size_t)b * DOUT + orow) * TLEN + tcol;
            #pragma unroll
            for (int r = 0; r < 8; ++r) obase[(size_t)r * TLEN] = acc[so][st][r];
        }
    }
}

// ---------------------------------------------------------------------------
extern "C" void kernel_launch(void* const* d_in, const int* in_sizes, int n_in,
                              void* d_out, int out_size, void* d_ws, size_t ws_size,
                              hipStream_t stream) {
    const float* x       = (const float*)d_in[0];
    const float* c_src   = (const float*)d_in[1];
    const float* c_trg   = (const float*)d_in[2];
    const float* style_w = (const float*)d_in[3];
    const float* style_b = (const float*)d_in[4];
    const float* weight  = (const float*)d_in[5];
    float* out = (float*)d_out;

    char* ws = (char*)d_ws;
    float* s  = (float*)ws;                                   // 16*512*4       = 32 KB
    bf16*  wb = (bf16*)(ws + 32768);                          // 16*3*512*512*2 = 24 MB
    bf16*  xT = (bf16*)(ws + 32768 + (size_t)NB * KW * DOUT * DIN * 2); // 32 MB

    // Phase 0: x -> bf16, transposed to [b][t][i]
    xpose_bf16<<<dim3(TLEN / 32, DIN / 32, NB), 256, 0, stream>>>(x, xT);
    // Phase 1: style codes
    style_kernel<<<dim3(2, NB), 256, 0, stream>>>(c_src, c_trg, style_w, style_b, s);
    // Phase 2: modulated/demodulated bf16 weights
    modw_kernel<<<dim3(DOUT, NB), 128, 0, stream>>>(weight, s, wb);
    // Phase 3: WMMA conv (block tile 128(O) x 128(T), 2x4 per wave)
    conv_wmma<<<dim3(TLEN / 128, DOUT / 128, NB), 256, 0, stream>>>(xT, wb, out);
}
